// Decoder_76974403879496
// MI455X (gfx1250) — matrix operand order repaired and checked by reading, not compiled, not run on hardware
//
#include <hip/hip_runtime.h>

typedef unsigned short u16;
typedef __attribute__((ext_vector_type(16))) __bf16 v16bf;
typedef __attribute__((ext_vector_type(16))) u16    v16u;
typedef __attribute__((ext_vector_type(8)))  u16    v8u;
typedef __attribute__((ext_vector_type(8)))  float  v8f;

#define LN_EPS 1e-5f
#define CELU_A 1.3f

// ---------- bf16 helpers (bit-pattern based, RNE-ish rounding) ----------
__device__ __forceinline__ u16 f2bfu(float f) {
  union { float f; unsigned u; } v; v.f = f;
  unsigned r = v.u + 0x7fffu + ((v.u >> 16) & 1u);
  return (u16)(r >> 16);
}
__device__ __forceinline__ float bfu2f(u16 h) {
  union { unsigned u; float f; } v; v.u = ((unsigned)h) << 16;
  return v.f;
}

// ---------- WMMA wrapper: D = A(16x32 bf16) x B(32x16 bf16) + C(f32) ----------
__device__ __forceinline__ v8f wmma_bf16(v16u a, v16u b, v8f c) {
  return __builtin_amdgcn_wmma_f32_16x16x32_bf16(
      false, __builtin_bit_cast(v16bf, a),
      false, __builtin_bit_cast(v16bf, b),
      (short)0, c, false, false);
}

// A-fragment (16-bit A 16x32): lane row = lane%16; element i -> K:
//   i<8 : half*8 + i ; i>=8 : 16 + half*8 + (i-8)
// => two contiguous 8-element (16B) loads at +half*8 and +16+half*8.
__device__ __forceinline__ v16u load_afrag(const u16* p, int hlf) {
  v8u lo = *(const v8u*)(p + hlf * 8);
  v8u hi = *(const v8u*)(p + 16 + hlf * 8);
  return __builtin_shufflevector(lo, hi, 0,1,2,3,4,5,6,7,8,9,10,11,12,13,14,15);
}
// B-fragment (16-bit B 32x16): lane col = lane%16; element i -> K:
//   i<8 : half*8 + i ; i>=8 : 16 + half*8 + (i-8)  (same as A)
// p points at Bt[col] row (transposed weight, contiguous in K).
__device__ __forceinline__ v16u load_bfrag(const u16* p, int hlf) {
  v8u lo = *(const v8u*)(p + hlf * 8);
  v8u hi = *(const v8u*)(p + 16 + hlf * 8);
  return __builtin_shufflevector(lo, hi, 0,1,2,3,4,5,6,7,8,9,10,11,12,13,14,15);
}
// elementwise bf16 multiply of two fragments (same element->K mapping)
__device__ __forceinline__ v16u mul_frag(v16u a, v16u b) {
  v16u r;
#pragma unroll
  for (int i = 0; i < 16; i++) r[i] = f2bfu(bfu2f(a[i]) * bfu2f(b[i]));
  return r;
}

// =====================================================================
// Generic bf16 WMMA GEMM:  out = epi(A(MxK) @ Bt^T + bias) [+resid]
// epi: 0=none, 1=relu, 2=celu+groupnorm (head-major output (B,8,S,64))
// Block: 256 thr = 8 waves; each wave does 32 rows x 64 cols (2 m-tiles x
// 4 n-tiles) so every B-fragment feeds 2 WMMAs. Grid: (M/256, N/64).
// =====================================================================
__global__ __launch_bounds__(256) void gemm_wmma_k(
    const u16* __restrict__ A, const u16* __restrict__ Bt,
    const float* __restrict__ bias, const float* __restrict__ resid,
    const float* __restrict__ gw, const float* __restrict__ gb,
    float* __restrict__ outF, u16* __restrict__ outBf,
    int N, int K, int S, int epi)
{
  int wave = threadIdx.x >> 5, lane = threadIdx.x & 31;
  int hlf = lane >> 4, l16 = lane & 15;
  int rBase = blockIdx.x * 256 + wave * 32;
  int cBase = blockIdx.y * 64;

  const u16* ap0 = A  + (size_t)(rBase +  0 + l16) * K;
  const u16* ap1 = A  + (size_t)(rBase + 16 + l16) * K;
  const u16* bp0 = Bt + (size_t)(cBase +  0 + l16) * K;
  const u16* bp1 = Bt + (size_t)(cBase + 16 + l16) * K;
  const u16* bp2 = Bt + (size_t)(cBase + 32 + l16) * K;
  const u16* bp3 = Bt + (size_t)(cBase + 48 + l16) * K;

  v8f acc[2][4] = {};
  const int PF = 256;  // prefetch distance in elements (512B ahead)
  for (int k0 = 0; k0 < K; k0 += 32) {
    // speculative prefetch of the K-stream (global_prefetch_b8)
    __builtin_prefetch(ap0 + k0 + PF, 0, 1);
    __builtin_prefetch(ap1 + k0 + PF, 0, 1);
    __builtin_prefetch(bp0 + k0 + PF, 0, 1);
    __builtin_prefetch(bp2 + k0 + PF, 0, 1);
    v16u a0 = load_afrag(ap0 + k0, hlf);
    v16u a1 = load_afrag(ap1 + k0, hlf);
    v16u b0 = load_bfrag(bp0 + k0, hlf);
    v16u b1 = load_bfrag(bp1 + k0, hlf);
    v16u b2 = load_bfrag(bp2 + k0, hlf);
    v16u b3 = load_bfrag(bp3 + k0, hlf);
    acc[0][0] = wmma_bf16(a0, b0, acc[0][0]);
    acc[0][1] = wmma_bf16(a0, b1, acc[0][1]);
    acc[0][2] = wmma_bf16(a0, b2, acc[0][2]);
    acc[0][3] = wmma_bf16(a0, b3, acc[0][3]);
    acc[1][0] = wmma_bf16(a1, b0, acc[1][0]);
    acc[1][1] = wmma_bf16(a1, b1, acc[1][1]);
    acc[1][2] = wmma_bf16(a1, b2, acc[1][2]);
    acc[1][3] = wmma_bf16(a1, b3, acc[1][3]);
  }

  int cols[4];
#pragma unroll
  for (int nt = 0; nt < 4; nt++) cols[nt] = cBase + nt * 16 + l16;

#pragma unroll
  for (int mt = 0; mt < 2; mt++) {
    float val[4][8];
#pragma unroll
    for (int nt = 0; nt < 4; nt++) {
      float bi = bias ? bias[cols[nt]] : 0.f;
#pragma unroll
      for (int r = 0; r < 8; r++) {
        float v = acc[mt][nt][r] + bi;
        if (epi == 1)      v = fmaxf(v, 0.f);
        else if (epi == 2) v = (v > 0.f) ? v : CELU_A * (__expf(v * (1.f / CELU_A)) - 1.f);
        val[nt][r] = v;
      }
    }
    int rB = rBase + mt * 16;
    if (epi == 2) {
      // group-norm over the 64 cols of this wave tile (== one head)
#pragma unroll
      for (int r = 0; r < 8; r++) {
        float s1 = val[0][r] + val[1][r] + val[2][r] + val[3][r];
        float s2 = val[0][r]*val[0][r] + val[1][r]*val[1][r]
                 + val[2][r]*val[2][r] + val[3][r]*val[3][r];
#pragma unroll
        for (int m = 1; m < 16; m <<= 1) {
          s1 += __shfl_xor(s1, m, 32);
          s2 += __shfl_xor(s2, m, 32);
        }
        float mu  = s1 * (1.f / 64.f);
        float var = s2 * (1.f / 64.f) - mu * mu;
        float rstd = rsqrtf(var + LN_EPS);
        int row = rB + r + 8 * hlf;
        int bb_ = row / S, ss = row - bb_ * S;
        size_t obase = (((size_t)bb_ * 8 + (size_t)(cBase >> 6)) * S + ss) * 64;
#pragma unroll
        for (int nt = 0; nt < 4; nt++) {
          float xn = (val[nt][r] - mu) * rstd * gw[cols[nt]] + gb[cols[nt]];
          int dh = nt * 16 + l16;
          if (outF)  outF[obase + dh]  = xn;
          if (outBf) outBf[obase + dh] = f2bfu(xn);
        }
      }
    } else {
#pragma unroll
      for (int r = 0; r < 8; r++) {
        int row = rB + r + 8 * hlf;
#pragma unroll
        for (int nt = 0; nt < 4; nt++) {
          float v = val[nt][r];
          size_t idx = (size_t)row * N + cols[nt];
          if (resid) v += resid[idx];
          if (outF)  outF[idx]  = v;
          if (outBf) outBf[idx] = f2bfu(v);
        }
      }
    }
  }
}

// =====================================================================
// Low-rank attention core. Grid (Sq/16, H=8, B=2), 256 threads (8 waves),
// each wave owns 2 query rows. Per q: amap = relu((q (.) K) @ Wb + bb) via
// WMMA, a_sp = masked-softmax(amap @ Wl + bl), pool = masked mean, then
// out = v1 * (a_sp @ V2) * sigmoid(pool @ Wl2 + bl2).
// =====================================================================
__global__ __launch_bounds__(256) void attn_core_k(
    const u16* __restrict__ Qbf, const u16* __restrict__ Kbf,
    const float* __restrict__ V1, const float* __restrict__ V2,
    const float* __restrict__ maskp,
    const u16* __restrict__ Wbt, const float* __restrict__ bb,
    const float* __restrict__ Wl, const float* __restrict__ bl,
    const float* __restrict__ Wl2, const float* __restrict__ bl2,
    float* __restrict__ Out, int Sq, int Sk)
{
  __shared__ float asp[16][264];
  __shared__ float poolsh[16][32];
  int wave = threadIdx.x >> 5, lane = threadIdx.x & 31;
  int hlf = lane >> 4, l16 = lane & 15;
  int b = blockIdx.z, h = blockIdx.y, bh = b * 8 + h;
  int qBase = blockIdx.x * 16;

  // Wb B-fragments (constant): Wbt is (32 x 64) bf16 (n-major, d contiguous)
  v16u wbf[2][2];
#pragma unroll
  for (int db = 0; db < 2; db++)
#pragma unroll
    for (int nt = 0; nt < 2; nt++)
      wbf[db][nt] = load_bfrag(Wbt + (size_t)(nt * 16 + l16) * 64 + db * 32, hlf);

  float bbv0 = bb[l16], bbv1 = bb[16 + l16];
  float wl0 = Wl[l16], wl1 = Wl[16 + l16];
  float bl0 = bl[0];

  for (int sub = 0; sub < 2; sub++) {
    int q = qBase + wave * 2 + sub;
    int slot = wave * 2 + sub;
    const u16* qp = Qbf + ((size_t)bh * Sq + q) * 64;
    v16u qf0 = load_afrag(qp, hlf);
    v16u qf1 = load_afrag(qp + 32, hlf);
    const float* mrow = maskp + ((size_t)b * Sq + q) * Sk;

    float pool0 = 0.f, pool1 = 0.f, cnt = 0.f;
    for (int kt = 0; kt < Sk / 16; kt++) {
      const u16* kp = Kbf + ((size_t)bh * Sk + kt * 16 + l16) * 64;
      __builtin_prefetch(kp + 1024, 0, 1);   // next k-tile rows
      v16u z0 = mul_frag(load_afrag(kp, hlf), qf0);
      v16u z1 = mul_frag(load_afrag(kp + 32, hlf), qf1);
      v8f c0 = {}, c1 = {};
      c0 = wmma_bf16(z0, wbf[0][0], c0);
      c0 = wmma_bf16(z1, wbf[1][0], c0);
      c1 = wmma_bf16(z0, wbf[0][1], c1);
      c1 = wmma_bf16(z1, wbf[1][1], c1);
#pragma unroll
      for (int r = 0; r < 8; r++) {
        int k = kt * 16 + r + 8 * hlf;
        float m = mrow[k];
        float t0 = fmaxf(c0[r] + bbv0, 0.f);
        float t1 = fmaxf(c1[r] + bbv1, 0.f);
        pool0 += m * t0; pool1 += m * t1; cnt += m;
        float sp = t0 * wl0 + t1 * wl1;   // partial over the 2x16 cols held by lanes
        sp += __shfl_xor(sp, 1, 32);
        sp += __shfl_xor(sp, 2, 32);
        sp += __shfl_xor(sp, 4, 32);
        sp += __shfl_xor(sp, 8, 32);
        if (l16 == 0) asp[slot][k] = (m == 0.f) ? -1e9f : (sp + bl0);
      }
    }
    cnt   += __shfl_xor(cnt, 16, 32);
    pool0 += __shfl_xor(pool0, 16, 32);
    pool1 += __shfl_xor(pool1, 16, 32);
    if (lane < 16) { poolsh[slot][lane] = pool0; poolsh[slot][16 + lane] = pool1; }

    // masked softmax over asp[slot][0..Sk)
    float mx = -1e30f;
    for (int i = lane; i < Sk; i += 32) mx = fmaxf(mx, asp[slot][i]);
#pragma unroll
    for (int m = 16; m >= 1; m >>= 1) mx = fmaxf(mx, __shfl_xor(mx, m, 32));
    float sum = 0.f;
    for (int i = lane; i < Sk; i += 32) {
      float e = __expf(asp[slot][i] - mx);
      asp[slot][i] = e; sum += e;
    }
#pragma unroll
    for (int m = 16; m >= 1; m >>= 1) sum += __shfl_xor(sum, m, 32);
    float inv = 1.f / sum;
    float invc = 1.f / cnt;

#pragma unroll
    for (int dd = 0; dd < 2; dd++) {
      int d = lane + dd * 32;
      const float* v2p = V2 + (size_t)bh * Sk * 64 + d;
      float acc = 0.f;
      for (int k = 0; k < Sk; k++) acc += asp[slot][k] * v2p[(size_t)k * 64];
      acc *= inv;
      float z = bl2[d];
#pragma unroll
      for (int n = 0; n < 32; n++) z += poolsh[slot][n] * invc * Wl2[n * 64 + d];
      float ach = 1.f / (1.f + __expf(-z));
      float v1v = V1[((size_t)bh * Sq + q) * 64 + d];
      Out[((size_t)b * Sq + q) * 512 + (size_t)h * 64 + d] = v1v * acc * ach;
    }
  }
}

// ---------- out = LN(A [+ Badd]); optionally writes sum and bf16 copy ----------
__global__ __launch_bounds__(256) void add_ln_k(
    const float* __restrict__ A, const float* __restrict__ Badd,
    const float* __restrict__ w, const float* __restrict__ bia,
    float* __restrict__ sumOut, float* __restrict__ lnOut, u16* __restrict__ lnBf)
{
  int row = blockIdx.x, tid = threadIdx.x;
  size_t base = (size_t)row * 512;
  float v0 = A[base + tid], v1 = A[base + tid + 256];
  if (Badd) { v0 += Badd[base + tid]; v1 += Badd[base + tid + 256]; }
  if (sumOut) { sumOut[base + tid] = v0; sumOut[base + tid + 256] = v1; }
  float s = v0 + v1, qq = v0 * v0 + v1 * v1;
#pragma unroll
  for (int m = 1; m < 32; m <<= 1) { s += __shfl_xor(s, m, 32); qq += __shfl_xor(qq, m, 32); }
  __shared__ float rs[8], rq[8];
  int wv = tid >> 5, lane = tid & 31;
  if (lane == 0) { rs[wv] = s; rq[wv] = qq; }
  __syncthreads();
  float S = 0.f, Q = 0.f;
#pragma unroll
  for (int i = 0; i < 8; i++) { S += rs[i]; Q += rq[i]; }
  float mu = S * (1.f / 512.f);
  float var = Q * (1.f / 512.f) - mu * mu;
  float r = rsqrtf(var + LN_EPS);
  float o0 = (v0 - mu) * r * w[tid] + bia[tid];
  float o1 = (v1 - mu) * r * w[tid + 256] + bia[tid + 256];
  lnOut[base + tid] = o0; lnOut[base + tid + 256] = o1;
  if (lnBf) { lnBf[base + tid] = f2bfu(o0); lnBf[base + tid + 256] = f2bfu(o1); }
}

// x2 = x1 + a2 ; xcbf = bf16[ x2 | word_x ]  (rows of 1024)
__global__ void add_pack_k(const float* __restrict__ x1, const float* __restrict__ a2,
                           const float* __restrict__ wx,
                           float* __restrict__ x2, u16* __restrict__ xcbf)
{
  int i = blockIdx.x * 256 + threadIdx.x;   // 512*512 exactly
  int row = i >> 9, col = i & 511;
  float v = x1[i] + a2[i];
  x2[i] = v;
  xcbf[(size_t)row * 1024 + col] = f2bfu(v);
  xcbf[(size_t)row * 1024 + 512 + col] = f2bfu(wx[i]);
}

__global__ void conv_bf_k(const float* __restrict__ in, u16* __restrict__ out, int n) {
  int i = blockIdx.x * 256 + threadIdx.x;
  if (i < n) out[i] = f2bfu(in[i]);
}
// in (K,N) f32 -> out (N,K) bf16
__global__ void trans_bf_k(const float* __restrict__ in, u16* __restrict__ out, int K, int N) {
  int i = blockIdx.x * 256 + threadIdx.x;
  if (i >= K * N) return;
  int n = i / K, k = i - n * K;
  out[i] = f2bfu(in[(size_t)k * N + n]);
}

// =====================================================================
extern "C" void kernel_launch(void* const* d_in, const int* in_sizes, int n_in,
                              void* d_out, int out_size, void* d_ws, size_t ws_size,
                              hipStream_t stream)
{
  (void)in_sizes; (void)n_in; (void)out_size; (void)ws_size;
  const float* x_in = (const float*)d_in[0];
  const float* enc  = (const float*)d_in[1];
  const float* seqm = (const float*)d_in[2];
  const float* attm = (const float*)d_in[3];

  char* wp = (char*)d_ws;
  auto alloc = [&](size_t bytes) -> void* {
    void* r = (void*)wp; wp += (bytes + 255) & ~(size_t)255; return r;
  };
  const int ND = 512;  // B*S rows (== B*M)

  u16*   xbf0   = (u16*)  alloc((size_t)ND * 512 * 2);
  u16*   encbf  = (u16*)  alloc((size_t)ND * 512 * 2);
  float* qf     = (float*)alloc((size_t)ND * 512 * 4);
  float* kf     = (float*)alloc((size_t)ND * 512 * 4);
  float* v1f    = (float*)alloc((size_t)ND * 512 * 4);
  float* v2f    = (float*)alloc((size_t)ND * 512 * 4);
  u16*   qbf    = (u16*)  alloc((size_t)ND * 512 * 2);
  u16*   kbf    = (u16*)  alloc((size_t)ND * 512 * 2);
  float* a1     = (float*)alloc((size_t)ND * 512 * 4);
  float* x1     = (float*)alloc((size_t)ND * 512 * 4);
  float* xn     = (float*)alloc((size_t)ND * 512 * 4);
  u16*   xnbf   = (u16*)  alloc((size_t)ND * 512 * 2);
  float* x2     = (float*)alloc((size_t)ND * 512 * 4);
  u16*   xcbf   = (u16*)  alloc((size_t)ND * 1024 * 2);
  float* x3     = (float*)alloc((size_t)ND * 512 * 4);
  float* x4     = (float*)alloc((size_t)ND * 512 * 4);
  u16*   x4bf   = (u16*)  alloc((size_t)ND * 512 * 2);
  u16*   hbf    = (u16*)  alloc((size_t)ND * 2048 * 2);
  float* x5     = (float*)alloc((size_t)ND * 512 * 4);
  float* xnext  = (float*)alloc((size_t)ND * 512 * 4);
  u16*   xnextbf= (u16*)  alloc((size_t)ND * 512 * 2);

  u16* W1T[2]; u16* W2T[2]; u16* WbiT[2];
  u16* projT[2][2][4]; u16* WbT[2][2];
  for (int L = 0; L < 2; L++) {
    W1T[L]  = (u16*)alloc((size_t)2048 * 512 * 2);
    W2T[L]  = (u16*)alloc((size_t)512 * 2048 * 2);
    WbiT[L] = (u16*)alloc((size_t)512 * 1024 * 2);
    for (int a = 0; a < 2; a++) {
      WbT[L][a] = (u16*)alloc(32 * 64 * 2);
      for (int j = 0; j < 4; j++) projT[L][a][j] = (u16*)alloc((size_t)512 * 512 * 2);
    }
  }

  auto conv = [&](const float* in, u16* out, int n) {
    conv_bf_k<<<dim3((n + 255) / 256), dim3(256), 0, stream>>>(in, out, n);
  };
  auto trans = [&](const float* in, u16* out, int K, int N) {
    trans_bf_k<<<dim3((K * N + 255) / 256), dim3(256), 0, stream>>>(in, out, K, N);
  };

  conv(x_in, xbf0, ND * 512);
  conv(enc, encbf, ND * 512);

  // ---- param pointer decode (JAX pytree: dicts flattened by sorted key) ----
  struct AttP {
    const float *Wb, *Wl, *Wl2, *bb, *bl, *bl2;
    const float *pW[4], *pb[4], *pgb[4], *pgw[4];   // order: k,q,v1,v2
  };
  struct LayP {
    const float *W1, *W2, *Wbi, *b1, *b2, *bbi;
    const float *lncw, *lncb, *lnfw, *lnfb, *lnxw, *lnxb;
    AttP att[2];  // 0=cross, 1=word
  };
  LayP lp[2];
  for (int L = 0; L < 2; L++) {
    int base = 4 + L * 56;
    lp[L].W1  = (const float*)d_in[base + 0];
    lp[L].W2  = (const float*)d_in[base + 1];
    lp[L].Wbi = (const float*)d_in[base + 2];
    lp[L].b1  = (const float*)d_in[base + 3];
    lp[L].b2  = (const float*)d_in[base + 4];
    lp[L].bbi = (const float*)d_in[base + 5];
    int ab[2] = { base + 6, base + 34 };   // cross block, word block
    for (int a = 0; a < 2; a++) {
      AttP& A_ = lp[L].att[a];
      int o = ab[a];
      A_.Wb  = (const float*)d_in[o + 0];
      A_.Wl  = (const float*)d_in[o + 1];
      A_.Wl2 = (const float*)d_in[o + 2];
      A_.bb  = (const float*)d_in[o + 3];
      A_.bl  = (const float*)d_in[o + 4];
      A_.bl2 = (const float*)d_in[o + 5];
      for (int j = 0; j < 4; j++) {      // k, q, v1, v2 each {W,b,gb,gw}
        A_.pW[j]  = (const float*)d_in[o + 6 + 4 * j];
        A_.pb[j]  = (const float*)d_in[o + 7 + 4 * j];
        A_.pgb[j] = (const float*)d_in[o + 8 + 4 * j];
        A_.pgw[j] = (const float*)d_in[o + 9 + 4 * j];
      }
    }
    lp[L].lncb = (const float*)d_in[base + 28];
    lp[L].lncw = (const float*)d_in[base + 29];
    lp[L].lnfb = (const float*)d_in[base + 30];
    lp[L].lnfw = (const float*)d_in[base + 31];
    lp[L].lnxb = (const float*)d_in[base + 32];
    lp[L].lnxw = (const float*)d_in[base + 33];

    trans(lp[L].W1, W1T[L], 512, 2048);
    trans(lp[L].W2, W2T[L], 2048, 512);
    trans(lp[L].Wbi, WbiT[L], 1024, 512);
    for (int a = 0; a < 2; a++) {
      trans(lp[L].att[a].Wb, WbT[L][a], 64, 32);
      for (int j = 0; j < 4; j++) trans(lp[L].att[a].pW[j], projT[L][a][j], 512, 512);
    }
  }

  auto gemm = [&](const u16* A, const u16* Bt, const float* bias, const float* resid,
                  const float* gw, const float* gb, float* oF, u16* oB,
                  int M, int N, int K, int epi) {
    gemm_wmma_k<<<dim3(M / 256, N / 64), dim3(256), 0, stream>>>(
        A, Bt, bias, resid, gw, gb, oF, oB, N, K, 256, epi);
  };
  auto attn = [&](const u16* Q, const u16* Kk, const float* V1p, const float* V2p,
                  const float* maskp, const AttP& A_, const u16* WbTp, float* outp) {
    attn_core_k<<<dim3(16, 8, 2), dim3(256), 0, stream>>>(
        Q, Kk, V1p, V2p, maskp, WbTp, A_.bb, A_.Wl, A_.bl, A_.Wl2, A_.bl2,
        outp, 256, 256);
  };

  const float* xcur = x_in;
  const u16* xbfc = xbf0;
  for (int L = 0; L < 2; L++) {
    LayP& P = lp[L];
    // ---------- word (self) low-rank attention ----------
    AttP& Aw = P.att[1];
    gemm(xbfc, projT[L][1][1], Aw.pb[1], nullptr, Aw.pgw[1], Aw.pgb[1], qf, qbf,  512, 512, 512, 2);
    gemm(xbfc, projT[L][1][0], Aw.pb[0], nullptr, Aw.pgw[0], Aw.pgb[0], kf, kbf,  512, 512, 512, 2);
    gemm(xbfc, projT[L][1][2], Aw.pb[2], nullptr, Aw.pgw[2], Aw.pgb[2], v1f, nullptr, 512, 512, 512, 2);
    gemm(xbfc, projT[L][1][3], Aw.pb[3], nullptr, Aw.pgw[3], Aw.pgb[3], v2f, nullptr, 512, 512, 512, 2);
    attn(qbf, kbf, v1f, v2f, seqm, Aw, WbT[L][1], a1);
    add_ln_k<<<dim3(512), dim3(256), 0, stream>>>(xcur, a1, P.lncw, P.lncb, x1, xn, xnbf);
    // ---------- cross low-rank attention ----------
    AttP& Ac = P.att[0];
    gemm(xnbf,  projT[L][0][1], Ac.pb[1], nullptr, Ac.pgw[1], Ac.pgb[1], qf, qbf,  512, 512, 512, 2);
    gemm(encbf, projT[L][0][0], Ac.pb[0], nullptr, Ac.pgw[0], Ac.pgb[0], kf, kbf,  512, 512, 512, 2);
    gemm(xnbf,  projT[L][0][2], Ac.pb[2], nullptr, Ac.pgw[2], Ac.pgb[2], v1f, nullptr, 512, 512, 512, 2);
    gemm(encbf, projT[L][0][3], Ac.pb[3], nullptr, Ac.pgw[3], Ac.pgb[3], v2f, nullptr, 512, 512, 512, 2);
    attn(qbf, kbf, v1f, v2f, attm, Ac, WbT[L][0], a1);  // a1 reused as a2
    // ---------- bottleneck + FFN ----------
    add_pack_k<<<dim3(1024), dim3(256), 0, stream>>>(x1, a1, xcur, x2, xcbf);
    gemm(xcbf, WbiT[L], P.bbi, xcur, nullptr, nullptr, x3, nullptr, 512, 512, 1024, 1);
    add_ln_k<<<dim3(512), dim3(256), 0, stream>>>(x3, nullptr, P.lnxw, P.lnxb, nullptr, x4, x4bf);
    gemm(x4bf, W1T[L], P.b1, nullptr, nullptr, nullptr, nullptr, hbf, 512, 2048, 512, 1);
    gemm(hbf,  W2T[L], P.b2, x4, nullptr, nullptr, x5, nullptr, 512, 512, 2048, 0);
    float* outp = (L == 1) ? (float*)d_out : xnext;
    add_ln_k<<<dim3(512), dim3(256), 0, stream>>>(x5, nullptr, P.lnfw, P.lnfb, nullptr, outp, xnextbf);
    xcur = outp; xbfc = xnextbf;
  }
}
